// GroupedQueryAttention_39805756899329
// MI455X (gfx1250) — compile-verified, matrix operand order checked by reading, not run on hardware
//
#include <hip/hip_runtime.h>
#include <hip/hip_bf16.h>
#include <stdint.h>

// ---------------- problem constants (from reference) ----------------
constexpr int kB  = 2;
constexpr int kS  = 2048;
constexpr int kE  = 2048;
constexpr int kH  = 16;
constexpr int kKV = 4;
constexpr int kHD = 128;
constexpr int kG  = kH / kKV;          // 4
constexpr int kM  = kB * kS;           // 4096 rows for all projections

typedef __bf16 bf16;
typedef bf16  v16bf __attribute__((ext_vector_type(16)));
typedef float v8f   __attribute__((ext_vector_type(8)));

// ---- gfx1250 async global->LDS copy (ASYNCcnt path), with safe fallback ----
#if defined(__has_builtin)
#if __has_builtin(__builtin_amdgcn_global_load_async_to_lds_b128) && \
    __has_builtin(__builtin_amdgcn_s_wait_asynccnt)
#define USE_ASYNC_LDS 1
#endif
#endif
#ifndef USE_ASYNC_LDS
#define USE_ASYNC_LDS 0
#endif

#define AS_GLOBAL __attribute__((address_space(1)))
#define AS_LDS    __attribute__((address_space(3)))

// Builtin expects gcc-style vector_size(16) int pointers (clang 'V4i*1'/'V4i*3').
typedef int v4i_vs __attribute__((vector_size(16)));

// Copy 16 bytes global -> LDS for this lane.
static __device__ __forceinline__ void cp16_g2l(const bf16* __restrict__ g,
                                                bf16* __restrict__ l) {
#if USE_ASYNC_LDS
    // Generic shared pointers carry the LDS byte offset in the low 32 bits.
    __builtin_amdgcn_global_load_async_to_lds_b128(
        (AS_GLOBAL v4i_vs*)(uintptr_t)g,
        (AS_LDS v4i_vs*)(uint32_t)(uintptr_t)l,
        0, 0);
#else
    *(uint4*)l = *(const uint4*)g;
#endif
}

static __device__ __forceinline__ void wait_async_lds() {
#if USE_ASYNC_LDS
    __builtin_amdgcn_s_wait_asynccnt(0);
#endif
}

static __device__ __forceinline__ v8f v8f_zero() {
    v8f z = {0.f, 0.f, 0.f, 0.f, 0.f, 0.f, 0.f, 0.f};
    return z;
}

// float -> bf16 with round-to-nearest-even, pure bit ops (no __bf16 arithmetic)
static __device__ __forceinline__ bf16 f2bf(float f) {
    union { float f; unsigned u; } a; a.f = f;
    unsigned r = a.u + 0x7FFFu + ((a.u >> 16) & 1u);
    union { unsigned short s; bf16 b; } o;
    o.s = (unsigned short)(r >> 16);
    return o.b;
}

static __device__ __forceinline__ v8f wmma_bf16(v16bf a, v16bf b, v8f c) {
    // (neg_a, A, neg_b, B, c_mod, C, reuse_a, reuse_b)
    return __builtin_amdgcn_wmma_f32_16x16x32_bf16(false, a, false, b, (short)0, c,
                                                   false, false);
}

union Frag { v16bf v; uint4 u[2]; };
union Oct  { uint4 u; bf16 e[8]; };

// ---------------- f32 -> bf16 convert ----------------
__global__ __launch_bounds__(256) void cvt_f32_bf16_kernel(const float* __restrict__ src,
                                                           bf16* __restrict__ dst,
                                                           size_t n) {
    size_t i = (size_t)blockIdx.x * blockDim.x + threadIdx.x;
    if (i < n) dst[i] = f2bf(src[i]);
}

// ---------------- RoPE (f32 in, bf16 out), interleaved-pair form ----------------
// src/dst layout: [(b*S+s), heads*HD], pair index i handles dims (2i, 2i+1)
__global__ __launch_bounds__(256) void rope_cvt_kernel(const float* __restrict__ src,
                                                       bf16* __restrict__ dst,
                                                       int heads, float scale, int total) {
    int idx = blockIdx.x * blockDim.x + threadIdx.x;
    if (idx >= total) return;
    int i    = idx & 63;           // pair 0..63
    int rest = idx >> 6;
    int hh   = rest % heads;
    int bs   = rest / heads;       // b*S + s
    int s    = bs & (kS - 1);
    float freq = __expf(-((float)(2 * i) / (float)kHD) * logf(10000.0f));
    float ang  = (float)s * freq;
    float sn, c;
    __sincosf(ang, &sn, &c);
    size_t base = ((size_t)bs * heads + hh) * kHD;
    float x1 = src[base + 2 * i];
    float x2 = src[base + 2 * i + 1];
    dst[base + 2 * i]     = f2bf((x1 * c - x2 * sn) * scale);
    dst[base + 2 * i + 1] = f2bf((x1 * sn + x2 * c) * scale);
}

// ---------------- bf16 WMMA GEMM: C(MxN) = A(MxK) * B(KxN) ----------------
// Block tile 128x128, 8 waves (2x4), each wave 64x32 via 4x2 wmma tiles, BK=32.
// OUT_BF16: 0 -> f32 output, 1 -> bf16 output.
constexpr int kAStr = 48;  // LDS row stride (bf16 elems), 96B: 16B-aligned, padded
constexpr int kBStr = 48;

template <int OUT_BF16>
__global__ __launch_bounds__(256)
void gemm_bf16_kernel(const bf16* __restrict__ A, const bf16* __restrict__ Bm,
                      void* __restrict__ Cout, int M, int N, int K) {
    __shared__ __align__(16) bf16 As[128 * kAStr];
    __shared__ __align__(16) bf16 Bs[128 * kBStr];

    const int tid    = threadIdx.x;
    const int wave   = tid >> 5;
    const int lane   = tid & 31;
    const int lhalf  = lane & 15;
    const int hi     = lane >> 4;          // which 16-lane half
    const int wave_m = wave >> 2;          // 0..1 -> 64 rows each
    const int wave_n = wave & 3;           // 0..3 -> 32 cols each
    const int m0     = blockIdx.y * 128;
    const int n0     = blockIdx.x * 128;
    const int k0     = hi * 8;             // per-half K sub-chunk base

    v8f acc[4][2];
    for (int a = 0; a < 4; ++a)
        for (int b = 0; b < 2; ++b) acc[a][b] = v8f_zero();

    for (int kt = 0; kt < K; kt += 32) {
        // Stage A tile: 128 rows x 32 cols, row-major in LDS (async 16B chunks)
        for (int i = tid; i < 128 * 4; i += 256) {
            int r = i >> 2, c = (i & 3) * 8;
            cp16_g2l(&A[(size_t)(m0 + r) * K + kt + c], &As[r * kAStr + c]);
        }
        // Stage B tile transposed: Bs[n][k] (column-of-B becomes contiguous K)
        for (int i = tid; i < 32 * 16; i += 256) {
            int kk = i >> 4, nc = (i & 15) * 8;
            Oct t;
            t.u = *(const uint4*)&Bm[(size_t)(kt + kk) * N + n0 + nc];
            #pragma unroll
            for (int j = 0; j < 8; ++j) Bs[(nc + j) * kBStr + kk] = t.e[j];
        }
        // Prefetch next k-tile while this one is consumed (global_prefetch_b8)
        if (kt + 32 < K) {
            __builtin_prefetch(&A[(size_t)(m0 + (tid >> 1)) * K + kt + 32], 0, 1);
            __builtin_prefetch(&Bm[(size_t)(kt + 32 + (tid >> 3)) * N + n0 + (tid & 7) * 16], 0, 1);
        }
        wait_async_lds();
        __syncthreads();

        // Fragments (16-bit A/B operand layout: lanes 0-15 K{0..7,16..23}, 16-31 K{8..15,24..31})
        v16bf afrag[4], bfrag[2];
        #pragma unroll
        for (int tm = 0; tm < 4; ++tm) {
            const bf16* p = &As[(wave_m * 64 + tm * 16 + lhalf) * kAStr];
            Frag t;
            t.u[0] = *(const uint4*)&p[k0];
            t.u[1] = *(const uint4*)&p[k0 + 16];
            afrag[tm] = t.v;
        }
        #pragma unroll
        for (int tn = 0; tn < 2; ++tn) {
            const bf16* p = &Bs[(wave_n * 32 + tn * 16 + lhalf) * kBStr];
            Frag t;
            t.u[0] = *(const uint4*)&p[k0];
            t.u[1] = *(const uint4*)&p[k0 + 16];
            bfrag[tn] = t.v;
        }
        #pragma unroll
        for (int tm = 0; tm < 4; ++tm)
            #pragma unroll
            for (int tn = 0; tn < 2; ++tn)
                acc[tm][tn] = wmma_bf16(afrag[tm], bfrag[tn], acc[tm][tn]);
        __syncthreads();
    }

    // Epilogue: C layout -> VGPR r is row (r + 8*hi), lane%16 is column
    #pragma unroll
    for (int tm = 0; tm < 4; ++tm)
        #pragma unroll
        for (int tn = 0; tn < 2; ++tn) {
            int mbase = m0 + wave_m * 64 + tm * 16 + hi * 8;
            int nn    = n0 + wave_n * 32 + tn * 16 + lhalf;
            #pragma unroll
            for (int r = 0; r < 8; ++r) {
                float v = acc[tm][tn][r];
                if (OUT_BF16)
                    ((bf16*)Cout)[(size_t)(mbase + r) * N + nn] = f2bf(v);
                else
                    ((float*)Cout)[(size_t)(mbase + r) * N + nn] = v;
            }
        }
}

// ---------------- Flash attention (causal, GQA), all-bf16 WMMA ----------------
// Grid: (S/64, H, B). 128 threads = 4 waves; wave w owns q-rows q0+16w..+15.
constexpr int kKsStr = 136;  // 32 keys x 128 dims, key-major rows (272B, 16B-aligned)
constexpr int kVsStr = 40;   // 128 dims x 32 keys, dim-major rows  (80B, 16B-aligned)
constexpr int kPsStr = 40;

__global__ __launch_bounds__(128)
void attn_kernel(const bf16* __restrict__ Qb, const bf16* __restrict__ Kb,
                 const bf16* __restrict__ Vb, bf16* __restrict__ Ob) {
    __shared__ __align__(16) bf16 Ks[32 * kKsStr];
    __shared__ __align__(16) bf16 Vs[128 * kVsStr];
    __shared__ __align__(16) bf16 Ps[4 * 16 * kPsStr];

    const int tid   = threadIdx.x;
    const int wave  = tid >> 5;
    const int lane  = tid & 31;
    const int lhalf = lane & 15;
    const int hi    = lane >> 4;
    const int q0    = blockIdx.x * 64;
    const int h     = blockIdx.y;
    const int b     = blockIdx.z;
    const int kvh   = h / kG;
    const int k0    = hi * 8;

    // Resident Q fragments: 16 rows x 128 dims = 4 k-steps of 32
    v16bf qfrag[4];
    {
        int row = q0 + wave * 16 + lhalf;
        const bf16* qp = &Qb[(((size_t)b * kS + row) * kH + h) * kHD];
        #pragma unroll
        for (int kk = 0; kk < 4; ++kk) {
            Frag t;
            t.u[0] = *(const uint4*)&qp[kk * 32 + k0];
            t.u[1] = *(const uint4*)&qp[kk * 32 + k0 + 16];
            qfrag[kk] = t.v;
        }
    }

    v8f oacc[8];
    #pragma unroll
    for (int t = 0; t < 8; ++t) oacc[t] = v8f_zero();
    float rmax[8], rsum[8];
    #pragma unroll
    for (int r = 0; r < 8; ++r) { rmax[r] = -3e30f; rsum[r] = 0.f; }

    const int kend = q0 + 64;
    for (int kt = 0; kt < kend; kt += 32) {
        // Stage K chunk key-major: Ks[key][d] (async global->LDS, no VGPR round-trip)
        for (int i = tid; i < 32 * 16; i += 128) {
            int key = i >> 4, c = (i & 15) * 8;
            cp16_g2l(&Kb[(((size_t)b * kS + kt + key) * kKV + kvh) * kHD + c],
                     &Ks[key * kKsStr + c]);
        }
        // Stage V chunk dim-major (transposed): Vs[d][key]
        for (int i = tid; i < 32 * 16; i += 128) {
            int key = i >> 4, c = (i & 15) * 8;
            Oct t;
            t.u = *(const uint4*)&Vb[(((size_t)b * kS + kt + key) * kKV + kvh) * kHD + c];
            #pragma unroll
            for (int j = 0; j < 8; ++j) Vs[(c + j) * kVsStr + key] = t.e[j];
        }
        if (kt + 32 < kend) {
            __builtin_prefetch(&Kb[(((size_t)b * kS + kt + 32 + (tid >> 2)) * kKV + kvh) * kHD], 0, 1);
            __builtin_prefetch(&Vb[(((size_t)b * kS + kt + 32 + (tid >> 2)) * kKV + kvh) * kHD], 0, 1);
        }
        wait_async_lds();
        __syncthreads();

        // Preload ALL K fragments, then issue the WMMA chain (avoids per-WMMA ds waits)
        v16bf kfa[4], kfb[4];
        #pragma unroll
        for (int kk = 0; kk < 4; ++kk) {
            Frag ta, tb;
            const bf16* kp0 = &Ks[(0 + lhalf) * kKsStr + kk * 32];
            ta.u[0] = *(const uint4*)&kp0[k0];
            ta.u[1] = *(const uint4*)&kp0[k0 + 16];
            const bf16* kp1 = &Ks[(16 + lhalf) * kKsStr + kk * 32];
            tb.u[0] = *(const uint4*)&kp1[k0];
            tb.u[1] = *(const uint4*)&kp1[k0 + 16];
            kfa[kk] = ta.v;
            kfb[kk] = tb.v;
        }
        v8f sa = v8f_zero(), sb = v8f_zero();
        #pragma unroll
        for (int kk = 0; kk < 4; ++kk) {
            sa = wmma_bf16(qfrag[kk], kfa[kk], sa);
            sb = wmma_bf16(qfrag[kk], kfb[kk], sb);
        }

        // Causal mask + online softmax (row r lives on this lane-half; 16-lane reduce)
        float pa[8], pb[8], rscale[8];
        #pragma unroll
        for (int r = 0; r < 8; ++r) {
            int m  = r + hi * 8;
            int qg = q0 + wave * 16 + m;
            float va = (kt + lhalf      <= qg) ? sa[r] : -3e30f;
            float vb = (kt + 16 + lhalf <= qg) ? sb[r] : -3e30f;
            float cm = fmaxf(va, vb);
            #pragma unroll
            for (int off = 1; off < 16; off <<= 1)
                cm = fmaxf(cm, __shfl_xor(cm, off, 32));
            float nm = fmaxf(rmax[r], cm);
            float ea = __expf(va - nm);
            float eb = __expf(vb - nm);
            float es = ea + eb;
            #pragma unroll
            for (int off = 1; off < 16; off <<= 1)
                es += __shfl_xor(es, off, 32);
            float sc  = __expf(rmax[r] - nm);
            rsum[r]   = rsum[r] * sc + es;
            rmax[r]   = nm;
            rscale[r] = sc;
            pa[r] = ea;
            pb[r] = eb;
        }
        #pragma unroll
        for (int t = 0; t < 8; ++t)
            #pragma unroll
            for (int r = 0; r < 8; ++r) oacc[t][r] *= rscale[r];

        // Repack P (16x32) into A-operand layout via per-wave LDS scratch
        bf16* pp = &Ps[wave * 16 * kPsStr];
        #pragma unroll
        for (int r = 0; r < 8; ++r) {
            int m = r + hi * 8;
            pp[m * kPsStr + lhalf]      = f2bf(pa[r]);
            pp[m * kPsStr + 16 + lhalf] = f2bf(pb[r]);
        }
        __syncthreads();
        Frag tp;
        const bf16* pr = &pp[lhalf * kPsStr];
        tp.u[0] = *(const uint4*)&pr[k0];
        tp.u[1] = *(const uint4*)&pr[k0 + 16];
        v16bf pfrag = tp.v;

        // Preload all 8 V fragments, then the PV WMMA chain
        v16bf vf[8];
        #pragma unroll
        for (int t = 0; t < 8; ++t) {
            Frag tv;
            const bf16* vp = &Vs[(t * 16 + lhalf) * kVsStr];
            tv.u[0] = *(const uint4*)&vp[k0];
            tv.u[1] = *(const uint4*)&vp[k0 + 16];
            vf[t] = tv.v;
        }
        #pragma unroll
        for (int t = 0; t < 8; ++t)
            oacc[t] = wmma_bf16(pfrag, vf[t], oacc[t]);
        __syncthreads();
    }

    // Normalize and store bf16 attention output, layout [(b*S+s), H*HD]
    #pragma unroll
    for (int r = 0; r < 8; ++r) {
        int m   = r + hi * 8;
        int qg  = q0 + wave * 16 + m;
        float inv = 1.0f / rsum[r];
        bf16* op = &Ob[(((size_t)b * kS + qg) * kH + h) * kHD];
        #pragma unroll
        for (int t = 0; t < 8; ++t) op[t * 16 + lhalf] = f2bf(oacc[t][r] * inv);
    }
}

// ---------------- host-side orchestration ----------------
extern "C" void kernel_launch(void* const* d_in, const int* in_sizes, int n_in,
                              void* d_out, int out_size, void* d_ws, size_t ws_size,
                              hipStream_t stream) {
    (void)in_sizes; (void)n_in; (void)out_size; (void)ws_size;
    const float* x  = (const float*)d_in[0];
    const float* Wq = (const float*)d_in[1];
    const float* Wk = (const float*)d_in[2];
    const float* Wv = (const float*)d_in[3];
    const float* Wo = (const float*)d_in[4];
    float* out = (float*)d_out;

    char* ws = (char*)d_ws;
    // Workspace layout (bytes). attnb aliases qf (qf dead after rope_q).
    const size_t n_x   = (size_t)kM * kE;          // 8,388,608
    const size_t n_q   = (size_t)kM * kH * kHD;    // 8,388,608
    const size_t n_kv  = (size_t)kM * kKV * kHD;   // 2,097,152
    size_t off = 0;
    bf16* xb  = (bf16*)(ws + off); off += n_x * 2;
    bf16* wqb = (bf16*)(ws + off); off += (size_t)kE * kH * kHD * 2;
    bf16* wkb = (bf16*)(ws + off); off += (size_t)kE * kKV * kHD * 2;
    bf16* wvb = (bf16*)(ws + off); off += (size_t)kE * kKV * kHD * 2;
    bf16* wob = (bf16*)(ws + off); off += (size_t)kH * kHD * kE * 2;
    bf16* qb  = (bf16*)(ws + off); off += n_q * 2;
    bf16* kb  = (bf16*)(ws + off); off += n_kv * 2;
    bf16* vb  = (bf16*)(ws + off); off += n_kv * 2;
    float* qf = (float*)(ws + off); off += n_q * 4;
    float* kf = (float*)(ws + off); off += n_kv * 4;
    bf16* attnb = (bf16*)qf;  // alias: 16.8MB into 33.5MB dead region

    auto cvt = [&](const float* s, bf16* d, size_t n) {
        cvt_f32_bf16_kernel<<<dim3((unsigned)((n + 255) / 256)), dim3(256), 0, stream>>>(s, d, n);
    };
    cvt(x,  xb,  n_x);
    cvt(Wq, wqb, (size_t)kE * kH * kHD);
    cvt(Wk, wkb, (size_t)kE * kKV * kHD);
    cvt(Wv, wvb, (size_t)kE * kKV * kHD);
    cvt(Wo, wob, (size_t)kH * kHD * kE);

    // Projections (M=4096, K=2048)
    gemm_bf16_kernel<0><<<dim3(kH * kHD / 128, kM / 128), dim3(256), 0, stream>>>(
        xb, wqb, (void*)qf, kM, kH * kHD, kE);
    gemm_bf16_kernel<0><<<dim3(kKV * kHD / 128, kM / 128), dim3(256), 0, stream>>>(
        xb, wkb, (void*)kf, kM, kKV * kHD, kE);
    gemm_bf16_kernel<1><<<dim3(kKV * kHD / 128, kM / 128), dim3(256), 0, stream>>>(
        xb, wvb, (void*)vb, kM, kKV * kHD, kE);

    // RoPE + bf16 (Q pre-scaled by 1/sqrt(HD) so scores are scaled)
    const float qscale = 0.08838834764831845f;  // 1/sqrt(128)
    {
        int totq = kM * kH * 64;
        rope_cvt_kernel<<<dim3(totq / 256), dim3(256), 0, stream>>>(qf, qb, kH, qscale, totq);
        int totk = kM * kKV * 64;
        rope_cvt_kernel<<<dim3(totk / 256), dim3(256), 0, stream>>>(kf, kb, kKV, 1.0f, totk);
    }

    // Causal GQA flash attention
    attn_kernel<<<dim3(kS / 64, kH, kB), dim3(128), 0, stream>>>(qb, kb, vb, attnb);

    // Output projection -> f32 d_out
    gemm_bf16_kernel<0><<<dim3(kE / 128, kM / 128), dim3(256), 0, stream>>>(
        attnb, wob, (void*)out, kM, kE, kH * kHD);
}
